// EncoderLayer_19404662243665
// MI455X (gfx1250) — matrix operand order repaired and checked by reading, not compiled, not run on hardware
//
#include <hip/hip_runtime.h>

// ---------------------------------------------------------------------------
// Transformer encoder layer for MI455X (gfx1250).
// bf16 WMMA (v_wmma_f32_16x16x32_bf16), 32x64 register-blocked macro-tiles:
// 8 WMMAs per 6KB of operand loads (~11 MAC/byte) vs 1 per 2KB unblocked.
// Shapes: S=2048, B=2, HD=1024, NH=16, DK=64, DFF=4096.
// ---------------------------------------------------------------------------

#define DEV __device__ __forceinline__

typedef __attribute__((ext_vector_type(16))) __bf16 v16bf;
typedef __attribute__((ext_vector_type(8)))  __bf16 v8bf;
typedef __attribute__((ext_vector_type(8)))  float  v8f;

static constexpr int S_ = 2048, B_ = 2, HD_ = 1024, NH_ = 16, DK_ = 64, DFF_ = 4096;

struct Frag32 { uint4 lo, hi; };   // 32 bytes == v16bf

DEV v8f wmma_bf16(v16bf a, v16bf b, v8f c) {
    return __builtin_amdgcn_wmma_f32_16x16x32_bf16(
        /*neg_a=*/false, a, /*neg_b=*/false, b,
        /*c_mod=*/(short)0, c, /*reuse_a=*/false, /*reuse_b=*/false);
}

// A fragment: 16x32 bf16, row-major source, stride lda (elements).
// Lane L: M = L&15, half = L>>4. dwords 0-3 = K half*8..+7, dwords 4-7 = K 16+half*8..+7.
DEV v16bf load_a_bf(const __bf16* A, int lda) {
    const int lane = threadIdx.x & 31;
    const int m = lane & 15, h = lane >> 4;
    const __bf16* p = A + (size_t)m * lda + h * 8;
    Frag32 f;
    f.lo = *reinterpret_cast<const uint4*>(p);
    f.hi = *reinterpret_cast<const uint4*>(p + 16);
    return __builtin_bit_cast(v16bf, f);
}

// B fragment: 32x16 bf16 supplied as Bt[N][K] row-major (B transposed), stride ldb.
// Lane L: N = L&15, kgroup = L>>4; dwords 0-3 = K kg*8..+7, dwords 4-7 = K 16+kg*8..+7.
DEV v16bf load_b_bf(const __bf16* Bt, int ldb) {
    const int lane = threadIdx.x & 31;
    const int n = lane & 15, kg = lane >> 4;
    const __bf16* p = Bt + (size_t)n * ldb + kg * 8;
    Frag32 f;
    f.lo = *reinterpret_cast<const uint4*>(p);
    f.hi = *reinterpret_cast<const uint4*>(p + 16);
    return __builtin_bit_cast(v16bf, f);
}

// A fragment from fp32 row-major source (att_w probabilities), cvt to bf16.
DEV v16bf load_a_f32(const float* A, int lda) {
    const int lane = threadIdx.x & 31;
    const int m = lane & 15, h = lane >> 4;
    const float* p = A + (size_t)m * lda + h * 8;
    const float4* q0 = reinterpret_cast<const float4*>(p);
    const float4* q1 = reinterpret_cast<const float4*>(p + 16);
    float4 f0 = q0[0], f1 = q0[1], f2 = q1[0], f3 = q1[1];
    v16bf a;
    a[0]=(__bf16)f0.x; a[1]=(__bf16)f0.y; a[2]=(__bf16)f0.z; a[3]=(__bf16)f0.w;
    a[4]=(__bf16)f1.x; a[5]=(__bf16)f1.y; a[6]=(__bf16)f1.z; a[7]=(__bf16)f1.w;
    a[8]=(__bf16)f2.x; a[9]=(__bf16)f2.y; a[10]=(__bf16)f2.z; a[11]=(__bf16)f2.w;
    a[12]=(__bf16)f3.x; a[13]=(__bf16)f3.y; a[14]=(__bf16)f3.z; a[15]=(__bf16)f3.w;
    return a;
}

// ---------------------------------------------------------------------------
// One-time conversion / layout kernels
// ---------------------------------------------------------------------------
__global__ void k_f32_to_bf16(const float* __restrict__ src, __bf16* __restrict__ dst, int n) {
    int i = blockIdx.x * blockDim.x + threadIdx.x;
    if (i < n) dst[i] = (__bf16)src[i];
}

// dst[z][c][r] = (bf16) src[z][r][c]   (batched weight transpose, W -> W^T)
__global__ void k_transpose_bf(const float* __restrict__ src, __bf16* __restrict__ dst,
                               int R, int C) {
    const size_t zoff = (size_t)blockIdx.y * R * C;
    int i = blockIdx.x * blockDim.x + threadIdx.x;
    if (i >= R * C) return;
    int r = i / C, c = i % C;
    dst[zoff + (size_t)c * R + r] = (__bf16)src[zoff + (size_t)r * C + c];
}

// ---------------------------------------------------------------------------
// QKV projection, 32x64 macro-tile per wave (covers full DK width).
// vmode==0: store row-major bf16 [z][s][d]  (Q, K)
// vmode==1: store transposed   bf16 [z][d][s]  (V -> Vt, packed 16B stores)
// ---------------------------------------------------------------------------
__global__ void k_proj(const __bf16* __restrict__ xbf, const __bf16* __restrict__ WT,
                       const float* __restrict__ bias, __bf16* __restrict__ out, int vmode) {
    const int z = blockIdx.z;                 // z = n*B + b
    const int n = z / B_, b = z % B_;
    const int s0 = blockIdx.x * 32;
    const int lda = B_ * HD_;
    const __bf16* A  = xbf + ((size_t)s0 * B_ + b) * HD_;     // row stride B*HD
    const __bf16* Bt = WT + (size_t)n * DK_ * HD_;
    v8f c[2][4] = {};
    for (int k = 0; k < HD_ / 32; ++k) {
        v16bf a0 = load_a_bf(A + k * 32, lda);
        v16bf a1 = load_a_bf(A + (size_t)16 * lda + k * 32, lda);
        #pragma unroll
        for (int j = 0; j < 4; ++j) {
            v16bf bb = load_b_bf(Bt + (size_t)j * 16 * HD_ + k * 32, HD_);
            c[0][j] = wmma_bf16(a0, bb, c[0][j]);
            c[1][j] = wmma_bf16(a1, bb, c[1][j]);
        }
    }
    const int lane = threadIdx.x & 31;
    const int ncol = lane & 15, mbase = (lane >> 4) * 8;
    if (vmode == 0) {
        __bf16* q = out + (size_t)z * S_ * DK_;
        #pragma unroll
        for (int i = 0; i < 2; ++i)
        #pragma unroll
        for (int j = 0; j < 4; ++j) {
            const int col = j * 16 + ncol;
            const float bv = bias[n * DK_ + col];
            #pragma unroll
            for (int r = 0; r < 8; ++r)
                q[(size_t)(s0 + i * 16 + mbase + r) * DK_ + col] = (__bf16)(c[i][j][r] + bv);
        }
    } else {
        #pragma unroll
        for (int i = 0; i < 2; ++i)
        #pragma unroll
        for (int j = 0; j < 4; ++j) {
            const int col = j * 16 + ncol;
            const float bv = bias[n * DK_ + col];
            v8bf t;
            #pragma unroll
            for (int r = 0; r < 8; ++r) t[r] = (__bf16)(c[i][j][r] + bv);
            __bf16* vt = out + (size_t)z * DK_ * S_ + (size_t)col * S_ + (s0 + i * 16 + mbase);
            *reinterpret_cast<uint4*>(vt) = __builtin_bit_cast(uint4, t);
        }
    }
}

// ---------------------------------------------------------------------------
// scores[z][s][t] = (Q_z[s,:] . K_z[t,:]) / 8 ; 32x64 macro-tile per wave.
// ---------------------------------------------------------------------------
__global__ void k_scores(const __bf16* __restrict__ Q, const __bf16* __restrict__ K,
                         float* __restrict__ att) {
    const int z = blockIdx.z;
    const int s0 = blockIdx.x * 32, t0 = blockIdx.y * 64;
    const __bf16* A  = Q + ((size_t)z * S_ + s0) * DK_;
    const __bf16* Bt = K + ((size_t)z * S_ + t0) * DK_;
    v8f c[2][4] = {};
    #pragma unroll
    for (int k = 0; k < DK_ / 32; ++k) {
        v16bf a0 = load_a_bf(A + k * 32, DK_);
        v16bf a1 = load_a_bf(A + (size_t)16 * DK_ + k * 32, DK_);
        #pragma unroll
        for (int j = 0; j < 4; ++j) {
            v16bf bb = load_b_bf(Bt + (size_t)j * 16 * DK_ + k * 32, DK_);
            c[0][j] = wmma_bf16(a0, bb, c[0][j]);
            c[1][j] = wmma_bf16(a1, bb, c[1][j]);
        }
    }
    const int lane = threadIdx.x & 31;
    const int ncol = lane & 15, mbase = (lane >> 4) * 8;
    #pragma unroll
    for (int i = 0; i < 2; ++i)
    #pragma unroll
    for (int j = 0; j < 4; ++j)
    #pragma unroll
    for (int r = 0; r < 8; ++r)
        att[((size_t)z * S_ + s0 + i * 16 + mbase + r) * S_ + t0 + j * 16 + ncol]
            = c[i][j][r] * 0.125f;
}

// ---------------------------------------------------------------------------
// In-place softmax over the last axis. One 256-thread block per row of 2048.
// ---------------------------------------------------------------------------
__global__ void k_softmax(float* __restrict__ att) {
    float* p = att + (size_t)blockIdx.x * S_;
    const int tid = threadIdx.x;
    float4* pv = reinterpret_cast<float4*>(p);
    float4 v0 = pv[tid * 2], v1 = pv[tid * 2 + 1];
    float m = fmaxf(fmaxf(fmaxf(v0.x, v0.y), fmaxf(v0.z, v0.w)),
                    fmaxf(fmaxf(v1.x, v1.y), fmaxf(v1.z, v1.w)));
    __shared__ float red[256];
    red[tid] = m; __syncthreads();
    for (int s = 128; s > 0; s >>= 1) {
        if (tid < s) red[tid] = fmaxf(red[tid], red[tid + s]);
        __syncthreads();
    }
    m = red[0]; __syncthreads();
    v0.x = __expf(v0.x - m); v0.y = __expf(v0.y - m);
    v0.z = __expf(v0.z - m); v0.w = __expf(v0.w - m);
    v1.x = __expf(v1.x - m); v1.y = __expf(v1.y - m);
    v1.z = __expf(v1.z - m); v1.w = __expf(v1.w - m);
    float s8 = v0.x + v0.y + v0.z + v0.w + v1.x + v1.y + v1.z + v1.w;
    red[tid] = s8; __syncthreads();
    for (int s = 128; s > 0; s >>= 1) {
        if (tid < s) red[tid] += red[tid + s];
        __syncthreads();
    }
    const float inv = 1.0f / red[0];
    v0.x *= inv; v0.y *= inv; v0.z *= inv; v0.w *= inv;
    v1.x *= inv; v1.y *= inv; v1.z *= inv; v1.w *= inv;
    pv[tid * 2] = v0; pv[tid * 2 + 1] = v1;
}

// ---------------------------------------------------------------------------
// attn[s,b, n*DK+d] = sum_t P_z[s,t] * V_z[t,d]; 32x64 macro-tile (full DK).
// P fp32 -> bf16 on the fly; Vt is [z][d][t] so B operand is contiguous.
// ---------------------------------------------------------------------------
__global__ void k_attn(const float* __restrict__ att, const __bf16* __restrict__ Vt,
                       __bf16* __restrict__ attnb) {
    const int z = blockIdx.z;
    const int n = z / B_, b = z % B_;
    const int s0 = blockIdx.x * 32;
    const float*  A  = att + ((size_t)z * S_ + s0) * S_;
    const __bf16* Bt = Vt + (size_t)z * DK_ * S_;
    v8f c[2][4] = {};
    for (int k = 0; k < S_ / 32; ++k) {
        v16bf a0 = load_a_f32(A + k * 32, S_);
        v16bf a1 = load_a_f32(A + (size_t)16 * S_ + k * 32, S_);
        #pragma unroll
        for (int j = 0; j < 4; ++j) {
            v16bf bb = load_b_bf(Bt + (size_t)j * 16 * S_ + k * 32, S_);
            c[0][j] = wmma_bf16(a0, bb, c[0][j]);
            c[1][j] = wmma_bf16(a1, bb, c[1][j]);
        }
    }
    const int lane = threadIdx.x & 31;
    const int ncol = lane & 15, mbase = (lane >> 4) * 8;
    #pragma unroll
    for (int i = 0; i < 2; ++i)
    #pragma unroll
    for (int j = 0; j < 4; ++j)
    #pragma unroll
    for (int r = 0; r < 8; ++r)
        attnb[((size_t)(s0 + i * 16 + mbase + r) * B_ + b) * HD_ + n * DK_ + j * 16 + ncol]
            = (__bf16)c[i][j][r];
}

// ---------------------------------------------------------------------------
// Generic GEMM: C[r,c] = A[r,:] . Bt[c,:] + bias[c]; optional ReLU;
// 32x64 macro-tile per wave; bf16 or fp32 row-major store, row stride N.
// ---------------------------------------------------------------------------
__global__ void k_gemm(const __bf16* __restrict__ A, const __bf16* __restrict__ Bt,
                       const float* __restrict__ bias,
                       __bf16* __restrict__ outb, float* __restrict__ outf,
                       int K, int N, int relu) {
    const int r0 = blockIdx.x * 32, c0 = blockIdx.y * 64;
    const __bf16* Ap = A  + (size_t)r0 * K;
    const __bf16* Bp = Bt + (size_t)c0 * K;
    v8f c[2][4] = {};
    const int ks = K / 32;
    for (int k = 0; k < ks; ++k) {
        if (k + 2 < ks) {
            __builtin_prefetch(Ap + (size_t)(k + 2) * 32, 0, 3);
            __builtin_prefetch(Bp + (size_t)(k + 2) * 32, 0, 3);
        }
        v16bf a0 = load_a_bf(Ap + k * 32, K);
        v16bf a1 = load_a_bf(Ap + (size_t)16 * K + k * 32, K);
        #pragma unroll
        for (int j = 0; j < 4; ++j) {
            v16bf bb = load_b_bf(Bp + (size_t)j * 16 * K + k * 32, K);
            c[0][j] = wmma_bf16(a0, bb, c[0][j]);
            c[1][j] = wmma_bf16(a1, bb, c[1][j]);
        }
    }
    const int lane = threadIdx.x & 31;
    const int ncol = lane & 15, mbase = (lane >> 4) * 8;
    #pragma unroll
    for (int i = 0; i < 2; ++i)
    #pragma unroll
    for (int j = 0; j < 4; ++j) {
        const int col = c0 + j * 16 + ncol;
        const float bv = bias[col];
        #pragma unroll
        for (int r = 0; r < 8; ++r) {
            float v = c[i][j][r] + bv;
            if (relu) v = fmaxf(v, 0.0f);
            if (outf) outf[(size_t)(r0 + i * 16 + mbase + r) * N + col] = v;
            else      outb[(size_t)(r0 + i * 16 + mbase + r) * N + col] = (__bf16)v;
        }
    }
}

// ---------------------------------------------------------------------------
// Host launch
// ---------------------------------------------------------------------------
extern "C" void kernel_launch(void* const* d_in, const int* in_sizes, int n_in,
                              void* d_out, int out_size, void* d_ws, size_t ws_size,
                              hipStream_t stream) {
    (void)in_sizes; (void)n_in; (void)out_size; (void)ws_size;
    const float* x  = (const float*)d_in[0];
    const float* Wq = (const float*)d_in[1];  const float* bq = (const float*)d_in[2];
    const float* Wk = (const float*)d_in[3];  const float* bk = (const float*)d_in[4];
    const float* Wv = (const float*)d_in[5];  const float* bv = (const float*)d_in[6];
    const float* Wo = (const float*)d_in[7];  const float* bo = (const float*)d_in[8];
    const float* W1 = (const float*)d_in[9];  const float* b1 = (const float*)d_in[10];
    const float* W2 = (const float*)d_in[11]; const float* b2 = (const float*)d_in[12];

    float* y_out = (float*)d_out;                       // [S,B,HD]
    float* att   = y_out + (size_t)S_ * B_ * HD_;       // [NH,B,S,S]

    // Workspace carve-out (~104 MB, all bf16)
    char* ws = (char*)d_ws;
    auto carve = [&](size_t bytes) -> __bf16* {
        __bf16* p = (__bf16*)ws;
        ws += (bytes + 255) & ~(size_t)255;
        return p;
    };
    __bf16* xbf  = carve((size_t)S_ * B_ * HD_ * 2);
    __bf16* WqT  = carve((size_t)NH_ * DK_ * HD_ * 2);
    __bf16* WkT  = carve((size_t)NH_ * DK_ * HD_ * 2);
    __bf16* WvT  = carve((size_t)NH_ * DK_ * HD_ * 2);
    __bf16* WoT  = carve((size_t)HD_ * HD_ * 2);
    __bf16* W1T  = carve((size_t)HD_ * DFF_ * 2);
    __bf16* W2T  = carve((size_t)DFF_ * HD_ * 2);
    __bf16* Qb   = carve((size_t)NH_ * B_ * S_ * DK_ * 2);
    __bf16* Kb   = carve((size_t)NH_ * B_ * S_ * DK_ * 2);
    __bf16* Vt   = carve((size_t)NH_ * B_ * DK_ * S_ * 2);
    __bf16* attnb= carve((size_t)S_ * B_ * HD_ * 2);
    __bf16* y1b  = carve((size_t)S_ * B_ * HD_ * 2);
    __bf16* hb   = carve((size_t)S_ * B_ * DFF_ * 2);

    // 1) convert x, transpose weights to bf16 [N][K]
    k_f32_to_bf16<<<(S_ * B_ * HD_) / 256, 256, 0, stream>>>(x, xbf, S_ * B_ * HD_);
    k_transpose_bf<<<dim3((HD_ * DK_) / 256, NH_), 256, 0, stream>>>(Wq, WqT, HD_, DK_);
    k_transpose_bf<<<dim3((HD_ * DK_) / 256, NH_), 256, 0, stream>>>(Wk, WkT, HD_, DK_);
    k_transpose_bf<<<dim3((HD_ * DK_) / 256, NH_), 256, 0, stream>>>(Wv, WvT, HD_, DK_);
    k_transpose_bf<<<dim3((HD_ * HD_) / 256, 1), 256, 0, stream>>>(Wo, WoT, HD_, HD_);
    k_transpose_bf<<<dim3((HD_ * DFF_) / 256, 1), 256, 0, stream>>>(W1, W1T, HD_, DFF_);
    k_transpose_bf<<<dim3((DFF_ * HD_) / 256, 1), 256, 0, stream>>>(W2, W2T, DFF_, HD_);

    // 2) Q/K/V projections (32x64 macro-tile per wave; V stored transposed)
    dim3 gproj(S_ / 32, 1, NH_ * B_);
    k_proj<<<gproj, 32, 0, stream>>>(xbf, WqT, bq, Qb, 0);
    k_proj<<<gproj, 32, 0, stream>>>(xbf, WkT, bk, Kb, 0);
    k_proj<<<gproj, 32, 0, stream>>>(xbf, WvT, bv, Vt, 1);

    // 3) scores -> att_w (fp32, to output), softmax in place
    k_scores<<<dim3(S_ / 32, S_ / 64, NH_ * B_), 32, 0, stream>>>(Qb, Kb, att);
    k_softmax<<<dim3(NH_ * B_ * S_), 256, 0, stream>>>(att);

    // 4) attn = P @ V  -> bf16 [s*B+b, HD]
    k_attn<<<dim3(S_ / 32, 1, NH_ * B_), 32, 0, stream>>>(att, Vt, attnb);

    // 5) y1 = attn @ Wo + bo ; h = relu(y1 @ W1 + b1) ; y = h @ W2 + b2
    k_gemm<<<dim3(S_ * B_ / 32, HD_ / 64), 32, 0, stream>>>(attnb, WoT, bo, y1b, nullptr, HD_, HD_, 0);
    k_gemm<<<dim3(S_ * B_ / 32, DFF_ / 64), 32, 0, stream>>>(y1b, W1T, b1, hb, nullptr, HD_, DFF_, 1);
    k_gemm<<<dim3(S_ * B_ / 32, HD_ / 64), 32, 0, stream>>>(hb, W2T, b2, nullptr, y_out, DFF_, HD_, 0);
}
